// MoELayer_2456721293915
// MI455X (gfx1250) — matrix operand order repaired and checked by reading, not compiled, not run on hardware
//
#include <hip/hip_runtime.h>
#include <hip/hip_bf16.h>

#define D_MODEL 1024
#define D_OUT   1024
#define NE      8
#define TOPK    2

#define BM 64
#define BN 128
#define BK 64
#define SA 80   // bf16 elems per A row in LDS (64 + 16 pad -> 160B stride)
#define SB 80   // bf16 elems per B col in LDS

typedef __attribute__((ext_vector_type(16))) __bf16 v16bf;
typedef __attribute__((ext_vector_type(8)))  __bf16 v8bf;
typedef __attribute__((ext_vector_type(4)))  __bf16 v4bf;
typedef __attribute__((ext_vector_type(8)))  float  v8f;

union AFrag { v16bf v; v8bf h[2]; };
union PK2  { __bf16 b[2]; unsigned int u; };

// ---------------------------------------------------------------- zero output
__global__ void k_zero(float4* __restrict__ out, long n4) {
    long i = (long)blockIdx.x * blockDim.x + threadIdx.x;
    long stride = (long)gridDim.x * blockDim.x;
    float4 z = {0.f, 0.f, 0.f, 0.f};
    for (; i < n4; i += stride) out[i] = z;
}

// ---------------------------------------------------------------- gating + top2
__global__ void k_gate(const float* __restrict__ x, const float* __restrict__ Wg,
                       const float* __restrict__ bg, int NT,
                       int* __restrict__ topE, float* __restrict__ topW) {
    __shared__ float sWg[NE * D_MODEL];
    for (int i = threadIdx.x; i < NE * D_MODEL; i += blockDim.x) sWg[i] = Wg[i];
    __syncthreads();

    const int wave = threadIdx.x >> 5;
    const int lane = threadIdx.x & 31;
    const int tok  = blockIdx.x * 8 + wave;
    if (tok >= NT) return;

    const float4* xv = (const float4*)(x + (long)tok * D_MODEL);
    float acc[NE];
#pragma unroll
    for (int e = 0; e < NE; e++) acc[e] = 0.f;

#pragma unroll
    for (int it = 0; it < D_MODEL / 128; it++) {
        const int k4 = it * 32 + lane;
        const float4 xr = xv[k4];
#pragma unroll
        for (int e = 0; e < NE; e++) {
            const float4 wr = ((const float4*)(sWg + e * D_MODEL))[k4];
            acc[e] += xr.x * wr.x + xr.y * wr.y + xr.z * wr.z + xr.w * wr.w;
        }
    }
#pragma unroll
    for (int e = 0; e < NE; e++) {
#pragma unroll
        for (int off = 16; off > 0; off >>= 1)
            acc[e] += __shfl_xor(acc[e], off, 32);
    }

    if (lane == 0) {
        float lg[NE];
#pragma unroll
        for (int e = 0; e < NE; e++) lg[e] = acc[e] + bg[e];
        int e1 = 0;
#pragma unroll
        for (int e = 1; e < NE; e++) if (lg[e] > lg[e1]) e1 = e;
        int e2 = (e1 == 0) ? 1 : 0;
#pragma unroll
        for (int e = 0; e < NE; e++) if (e != e1 && lg[e] > lg[e2]) e2 = e;
        const float w1 = 1.f / (1.f + expf(-lg[e1]));
        const float w2 = 1.f / (1.f + expf(-lg[e2]));
        topE[tok * 2 + 0] = e1;  topW[tok * 2 + 0] = w1;
        topE[tok * 2 + 1] = e2;  topW[tok * 2 + 1] = w2;
    }
}

// ------------------------------------------------- chunked histogram (256 assigns/chunk)
__global__ void k_chunk_count(const int* __restrict__ topE, int nAssign,
                              int* __restrict__ chunkCounts) {
    __shared__ int h[NE];
    if (threadIdx.x < NE) h[threadIdx.x] = 0;
    __syncthreads();
    const int a = blockIdx.x * 256 + threadIdx.x;
    if (a < nAssign) atomicAdd(&h[topE[a]], 1);
    __syncthreads();
    if (threadIdx.x < NE) chunkCounts[blockIdx.x * NE + threadIdx.x] = h[threadIdx.x];
}

// ------------------------------------------------- prefix sums (deterministic)
__global__ void k_prefix(const int* __restrict__ chunkCounts, int C,
                         int* __restrict__ chunkBase, int* __restrict__ count,
                         int* __restrict__ segBase) {
    __shared__ int tot[NE];
    const int e = threadIdx.x;
    if (e < NE) {
        int run = 0;
        for (int c = 0; c < C; c++) { chunkBase[c * NE + e] = run; run += chunkCounts[c * NE + e]; }
        tot[e] = run;  count[e] = run;
    }
    __syncthreads();
    if (threadIdx.x == 0) {
        int run = 0;
        for (int q = 0; q < NE; q++) { segBase[q] = run; run += tot[q]; }
    }
    __syncthreads();
    if (e < NE) {
        const int b = segBase[e];
        for (int c = 0; c < C; c++) chunkBase[c * NE + e] += b;
    }
}

// ------------------------------------------------- stable scatter (per-chunk sequential)
__global__ void k_scatter(const int* __restrict__ topE, const float* __restrict__ topW,
                          int nAssign, const int* __restrict__ chunkBase,
                          int* __restrict__ tokList, float* __restrict__ wList) {
    if (threadIdx.x != 0) return;
    const int c = blockIdx.x;
    int cur[NE];
#pragma unroll
    for (int e = 0; e < NE; e++) cur[e] = chunkBase[c * NE + e];
    const int a0 = c * 256;
    const int aEnd = (a0 + 256 < nAssign) ? a0 + 256 : nAssign;
    for (int a = a0; a < aEnd; a++) {
        const int e = topE[a];
        const int p = cur[e]++;
        tokList[p] = a >> 1;
        wList[p]   = topW[a];
    }
}

// ------------------------------------------------- per-expert gathered GEMM (bf16 WMMA)
// block tile 64x128, 8 waves, each wave owns 16 rows x 64 cols (4 accumulators)
__global__ __launch_bounds__(256)
void k_moe_gemm(const float* __restrict__ x, const float* __restrict__ We,
                const float* __restrict__ be, const int* __restrict__ count,
                const int* __restrict__ segBase, const int* __restrict__ tokList,
                const float* __restrict__ wList, float* __restrict__ out) {
    const int e   = blockIdx.z;
    const int cnt = count[e];
    const int m0  = blockIdx.y * BM;
    if (m0 >= cnt) return;                       // uniform early-exit
    const int n0   = blockIdx.x * BN;
    const int base = segBase[e];

    __shared__ __align__(32) __bf16 aT[BM * SA]; // [m][k]
    __shared__ __align__(32) __bf16 bT[BN * SB]; // [n][k] (transposed)
    __shared__ int   sTok[BM];
    __shared__ float sW[BM];

    const int tid = threadIdx.x;
    if (tid < BM) {
        const int m = m0 + tid;
        const bool v = m < cnt;
        sTok[tid] = v ? tokList[base + m] : 0;
        sW[tid]   = v ? wList[base + m] : 0.f;
    }
    __syncthreads();

    const int lane = tid & 31;
    const int wv   = tid >> 5;       // 8 waves
    const int wm   = wv >> 1;        // 0..3 : 16-row group
    const int wn   = wv & 1;         // 0..1 : 64-col group

    v8f cacc[4] = {v8f{}, v8f{}, v8f{}, v8f{}};

    // A staging: 4 threads/row, each 16 floats
    const int arow = tid >> 2;
    const int acol = (tid & 3) * 16;
    const float* aRowPtr = x + (long)sTok[arow] * D_MODEL;
    // B staging: thread handles 2 consecutive k-rows x 16 cols (pairs pack to b32 stores)
    const int krow  = (tid >> 3) * 2;       // 0,2,..,62
    const int col16 = (tid & 7) * 16;       // 0..112
    const float* bBase = We + (long)e * D_MODEL * D_OUT + n0 + col16;

    float4 aR[4], bR[8];

    // prologue: load K-slab 0 into registers
    {
        const float4* ap = (const float4*)(aRowPtr + acol);
#pragma unroll
        for (int q = 0; q < 4; q++) aR[q] = ap[q];
#pragma unroll
        for (int r = 0; r < 2; r++) {
            const float4* bp = (const float4*)(bBase + (long)(krow + r) * D_OUT);
#pragma unroll
            for (int q = 0; q < 4; q++) bR[r * 4 + q] = bp[q];
        }
    }

    for (int k0 = 0; k0 < D_MODEL; k0 += BK) {
        // convert + store staged registers into LDS
#pragma unroll
        for (int q = 0; q < 4; q++) {
            v4bf pa;
            pa[0] = (__bf16)aR[q].x; pa[1] = (__bf16)aR[q].y;
            pa[2] = (__bf16)aR[q].z; pa[3] = (__bf16)aR[q].w;
            *(v4bf*)&aT[arow * SA + acol + q * 4] = pa;   // 8B ds stores
        }
#pragma unroll
        for (int q = 0; q < 4; q++) {
            const float4 lo = bR[q], hi = bR[4 + q];
            const float l4[4] = {lo.x, lo.y, lo.z, lo.w};
            const float h4[4] = {hi.x, hi.y, hi.z, hi.w};
#pragma unroll
            for (int i = 0; i < 4; i++) {
                PK2 pk;
                pk.b[0] = (__bf16)l4[i];      // k = krow
                pk.b[1] = (__bf16)h4[i];      // k = krow+1
                const int n = col16 + q * 4 + i;
                *(unsigned int*)&bT[n * SB + krow] = pk.u;  // 4B ds stores
            }
        }
        __syncthreads();

        // issue next slab's global loads before computing (overlap with WMMA)
        const int kn = k0 + BK;
        if (kn < D_MODEL) {
            const float4* ap = (const float4*)(aRowPtr + kn + acol);
#pragma unroll
            for (int q = 0; q < 4; q++) aR[q] = ap[q];
#pragma unroll
            for (int r = 0; r < 2; r++) {
                const float4* bp = (const float4*)(bBase + (long)(kn + krow + r) * D_OUT);
#pragma unroll
                for (int q = 0; q < 4; q++) bR[r * 4 + q] = bp[q];
            }
        }

#pragma unroll
        for (int ks = 0; ks < BK / 32; ks++) {
            AFrag af;
            const __bf16* ap = &aT[(wm * 16 + (lane & 15)) * SA + ks * 32];
            af.v = *(const v16bf*)(ap + 16 * (lane >> 4));
#pragma unroll
            for (int j = 0; j < 4; j++) {
                const v16bf bf = *(const v16bf*)&bT[(wn * 64 + j * 16 + (lane & 15)) * SB
                                                   + ks * 32 + 16 * (lane >> 4)];
                cacc[j] = __builtin_amdgcn_wmma_f32_16x16x32_bf16(
                    false, af.v, false, bf, (short)0, cacc[j], false, false);
            }
        }
        __syncthreads();
    }

    // epilogue: weight by gate, add bias, hardware f32 atomics (2 contribs/token)
    const int mBaseW = wm * 16 + ((lane >> 4) << 3);
    const int nl = lane & 15;
#pragma unroll
    for (int j = 0; j < 4; j++) {
        const int nG = n0 + wn * 64 + j * 16 + nl;
        const float bias = be[e * D_OUT + nG];
#pragma unroll
        for (int i = 0; i < 8; i++) {
            const int ml  = mBaseW + i;
            const int tok = sTok[ml];
            const float w = sW[ml];
            unsafeAtomicAdd(&out[(long)tok * D_OUT + nG], (cacc[j][i] + bias) * w);
        }
    }
}

// ---------------------------------------------------------------- launch
extern "C" void kernel_launch(void* const* d_in, const int* in_sizes, int n_in,
                              void* d_out, int out_size, void* d_ws, size_t ws_size,
                              hipStream_t stream) {
    const float* x  = (const float*)d_in[0];
    const float* Wg = (const float*)d_in[1];
    const float* bg = (const float*)d_in[2];
    const float* We = (const float*)d_in[3];
    const float* be = (const float*)d_in[4];
    float* out = (float*)d_out;

    const int NT = in_sizes[0] / D_MODEL;
    const int nAssign = NT * TOPK;
    const int C = (nAssign + 255) / 256;

    char* w = (char*)d_ws;
    int*   topE        = (int*)w;   w += sizeof(int)   * (size_t)nAssign;
    float* topW        = (float*)w; w += sizeof(float) * (size_t)nAssign;
    int*   chunkCounts = (int*)w;   w += sizeof(int)   * (size_t)C * NE;
    int*   chunkBase   = (int*)w;   w += sizeof(int)   * (size_t)C * NE;
    int*   count       = (int*)w;   w += sizeof(int)   * NE;
    int*   segBase     = (int*)w;   w += sizeof(int)   * NE;
    int*   tokList     = (int*)w;   w += sizeof(int)   * (size_t)nAssign;
    float* wList       = (float*)w; w += sizeof(float) * (size_t)nAssign;

    k_zero<<<2048, 256, 0, stream>>>((float4*)out, (long)out_size / 4);
    k_gate<<<(NT + 7) / 8, 256, 0, stream>>>(x, Wg, bg, NT, topE, topW);
    k_chunk_count<<<C, 256, 0, stream>>>(topE, nAssign, chunkCounts);
    k_prefix<<<1, 32, 0, stream>>>(chunkCounts, C, chunkBase, count, segBase);
    k_scatter<<<C, 32, 0, stream>>>(topE, topW, nAssign, chunkBase, tokList, wList);

    dim3 g(D_OUT / BN, (nAssign + BM - 1) / BM, NE);
    k_moe_gemm<<<g, 256, 0, stream>>>(x, We, be, count, segBase, tokList, wList, out);
}
